// SpatialTemporalAttention_53584011985562
// MI455X (gfx1250) — matrix operand order repaired and checked by reading, not compiled, not run on hardware
//
#include <hip/hip_runtime.h>

#define BDIM 512
#define TDIM 256
#define HDIM 256
#define TT   64          // T-tile rows
#define LDH  264         // padded LDS row stride (bf16 elems): 528B -> 4-bank shift/row

typedef __attribute__((ext_vector_type(16))) __bf16 v16bf;
typedef __attribute__((ext_vector_type(8)))  float  v8f;

union Frag {
    v16bf v;
    uint4 u[2];
};

__device__ __forceinline__ unsigned short f2bf(float f) {
    unsigned int x = __float_as_uint(f);
    unsigned int r = (x + 0x7FFFu + ((x >> 16) & 1u)) >> 16;   // RNE
    return (unsigned short)r;
}

// ---------------------------------------------------------------------------
// Prep 1: convert W_Ih / W_oI (fp32 [H][H], row = output o, col = contraction k)
// into bf16 B-matrix fragments for v_wmma_f32_16x16x32_bf16.
// Block id = bk*16 + bn (bk = k0/32, bn = n0/16). Lane L holds
// K = k0 + (L>=16?16:0) + e (e=0..15, contiguous) for N = n0+(L&15),
// stored contiguously so each lane issues two global_load_b128.
// ---------------------------------------------------------------------------
__global__ __launch_bounds__(256)
void convert_weights(const float* __restrict__ W_Ih,
                     const float* __restrict__ W_oI,
                     unsigned short* __restrict__ wb1,
                     unsigned short* __restrict__ wb2) {
    int gid = blockIdx.x * 256 + threadIdx.x;       // 0 .. 2*65536-1
    int mat = gid >> 16;
    int idx = gid & 65535;                          // == blk*512 + L*16 + e
    int e   = idx & 15;
    int L   = (idx >> 4) & 31;
    int blk = idx >> 9;
    int bn  = blk & 15;
    int bk  = blk >> 4;
    int k   = bk * 32 + ((L & 16) ? 16 : 0) + e;
    int n   = bn * 16 + (L & 15);
    const float* src = (mat == 0) ? W_Ih : W_oI;
    unsigned short* dst = (mat == 0) ? wb1 : wb2;
    dst[idx] = f2bf(src[n * HDIM + k]);             // B[k][n] = W[o=n][h=k]
}

// ---------------------------------------------------------------------------
// Prep 2: current_proj[b][o] = [h_tilde|c_t] @ W_I.T + b_I   (tiny GEMV)
// ---------------------------------------------------------------------------
__global__ __launch_bounds__(256)
void current_proj_kernel(const float* __restrict__ h_tilde,
                         const float* __restrict__ c_t,
                         const float* __restrict__ W_I,
                         const float* __restrict__ b_I,
                         float* __restrict__ cp) {
    __shared__ float hh[2 * HDIM];
    const int b = blockIdx.x, o = threadIdx.x;
    hh[o]        = h_tilde[b * HDIM + o];
    hh[o + HDIM] = c_t[b * HDIM + o];
    __syncthreads();
    float a = b_I[o];
    const float4* wr  = (const float4*)(W_I + (size_t)o * 2 * HDIM);
    const float4* hh4 = (const float4*)hh;
#pragma unroll 8
    for (int k = 0; k < 128; ++k) {
        float4 w = wr[k];
        float4 h = hh4[k];
        a += w.x * h.x + w.y * h.y + w.z * h.z + w.w * h.w;
    }
    cp[b * HDIM + o] = a;
}

// ---------------------------------------------------------------------------
// Wave-level GEMM over one 64x256 A-tile in LDS against 256x256 swizzled B.
// 8 waves: wave w owns column tiles {2w, 2w+1}; 4 row tiles each.
// K-loop kept ROLLED (unroll 1) to bound register pressure (~100 VGPRs:
// 64 acc + 16 B + 8 A) -> no spills, no vgpr_msb traffic. Each A-fragment
// is loaded once and feeds both column tiles.
// ---------------------------------------------------------------------------
__device__ __forceinline__ void gemm_tile(const unsigned short* __restrict__ wsw,
                                          const unsigned short* hbuf,
                                          int wave, int lane, int l15, int hsel,
                                          v8f acc[2][4]) {
    const int cb0 = wave * 2;
#pragma unroll 1
    for (int kb = 0; kb < 8; ++kb) {
        Frag B0, B1;
        const uint4* bp0 = (const uint4*)(wsw + ((kb * 16 + cb0) * 32 + lane) * 16);
        const uint4* bp1 = (const uint4*)(wsw + ((kb * 16 + cb0 + 1) * 32 + lane) * 16);
        B0.u[0] = bp0[0];
        B0.u[1] = bp0[1];
        B1.u[0] = bp1[0];
        B1.u[1] = bp1[1];
#pragma unroll
        for (int rt = 0; rt < 4; ++rt) {
            Frag A;
            const int m = rt * 16 + l15;
            A.u[0] = *(const uint4*)(&hbuf[m * LDH + kb * 32 + 2 * hsel]);
            A.u[1] = *(const uint4*)(&hbuf[m * LDH + kb * 32 + 2 * hsel + 8]);
            acc[0][rt] = __builtin_amdgcn_wmma_f32_16x16x32_bf16(
                false, A.v, false, B0.v, (short)0, acc[0][rt], false, false);
            acc[1][rt] = __builtin_amdgcn_wmma_f32_16x16x32_bf16(
                false, A.v, false, B1.v, (short)0, acc[1][rt], false, false);
        }
    }
}

// ---------------------------------------------------------------------------
// Fused attention kernel: one workgroup per batch element.
// ---------------------------------------------------------------------------
__global__ __launch_bounds__(256)
void attn_fused(const float* __restrict__ h_hist,
                const float* __restrict__ dT_g,
                const float* __restrict__ dL_g,
                const float* __restrict__ W_I1T,
                const float* __restrict__ W_I1L,
                const float* __restrict__ b_oI,
                const float* __restrict__ v_t,
                const unsigned short* __restrict__ wb1,
                const unsigned short* __restrict__ wb2,
                const float* __restrict__ cp,
                float* __restrict__ out_s,
                float* __restrict__ out_alpha) {
    __shared__ __align__(16) unsigned short hbuf[TT * LDH];  // 33 KB: h tile, then I tile
    __shared__ float scores_all[TDIM];
    __shared__ float stile[TT];
    __shared__ float dT[TT], dL[TT];
    __shared__ float cpb[HDIM], wTl[HDIM], wLl[HDIM], bol[HDIM], vvl[HDIM];
    __shared__ float red[8];
    __shared__ float gscal[2];

    const int b    = blockIdx.x;
    const int tid  = threadIdx.x;
    const int wave = tid >> 5;
    const int lane = tid & 31;
    const int l15  = lane & 15;
    const int hsel = (lane < 16) ? 0 : 8;

    cpb[tid] = cp[b * HDIM + tid];
    wTl[tid] = W_I1T[tid];
    wLl[tid] = W_I1L[tid];
    bol[tid] = b_oI[tid];
    vvl[tid] = v_t[tid];
    __syncthreads();

    const v8f vz = {0.f, 0.f, 0.f, 0.f, 0.f, 0.f, 0.f, 0.f};

#pragma unroll 1
    for (int tile = 0; tile < 4; ++tile) {
        const int t0 = tile * TT;
        if (tid < TT) {
            stile[tid] = 0.f;
            dT[tid] = dT_g[b * TDIM + t0 + tid];
            dL[tid] = dL_g[b * TDIM + t0 + tid];
        }
        if (tile < 3) {  // prefetch next h tile (global_prefetch_b8)
            __builtin_prefetch(h_hist + (size_t)b * TDIM * HDIM + (size_t)(t0 + TT) * HDIM + tid * 64, 0, 1);
        }
        // ---- stage h_history tile -> bf16 LDS ----
        const float* hb = h_hist + (size_t)b * TDIM * HDIM + (size_t)t0 * HDIM;
#pragma unroll 4
        for (int i = 0; i < 16; ++i) {
            int q  = tid + i * 256;      // 0..4095 float4-chunks
            int r  = q >> 6;
            int c4 = q & 63;
            float4 f = *(const float4*)(hb + r * HDIM + c4 * 4);
            unsigned short* d = &hbuf[r * LDH + c4 * 4];
            d[0] = f2bf(f.x); d[1] = f2bf(f.y); d[2] = f2bf(f.z); d[3] = f2bf(f.w);
        }
        __syncthreads();

        // ---- GEMM1: M1 = h_tile @ W_Ih^T ----
        v8f acc[2][4];
#pragma unroll
        for (int cj = 0; cj < 2; ++cj)
#pragma unroll
            for (int rt = 0; rt < 4; ++rt) acc[cj][rt] = vz;
        gemm_tile(wb1, hbuf, wave, lane, l15, hsel, acc);
        __syncthreads();  // all waves done reading h tile

        // ---- epilogue 1: I = M1 + cp + dT*wT + dL*wL  (bf16, overwrite hbuf) ----
#pragma unroll
        for (int cj = 0; cj < 2; ++cj) {
            const int col = (wave * 2 + cj) * 16 + l15;
            const float cpv = cpb[col], wtv = wTl[col], wlv = wLl[col];
#pragma unroll
            for (int rt = 0; rt < 4; ++rt) {
                const int rowb = rt * 16 + hsel;
#pragma unroll
                for (int r = 0; r < 8; ++r) {
                    const int row = rowb + r;
                    float v = acc[cj][rt][r] + cpv + dT[row] * wtv + dL[row] * wlv;
                    hbuf[row * LDH + col] = f2bf(v);
                }
            }
        }
        __syncthreads();

        // ---- GEMM2: O_pre = I_tile @ W_oI^T ----
#pragma unroll
        for (int cj = 0; cj < 2; ++cj)
#pragma unroll
            for (int rt = 0; rt < 4; ++rt) acc[cj][rt] = vz;
        gemm_tile(wb2, hbuf, wave, lane, l15, hsel, acc);

        // ---- epilogue 2: scores[t] += tanh(O_pre + b_oI) . v_t ----
        float sp[4][8];
#pragma unroll
        for (int rt = 0; rt < 4; ++rt)
#pragma unroll
            for (int r = 0; r < 8; ++r) sp[rt][r] = 0.f;
#pragma unroll
        for (int cj = 0; cj < 2; ++cj) {
            const int col = (wave * 2 + cj) * 16 + l15;
            const float bov = bol[col], vvv = vvl[col];
#pragma unroll
            for (int rt = 0; rt < 4; ++rt)
#pragma unroll
                for (int r = 0; r < 8; ++r)
                    sp[rt][r] += tanhf(acc[cj][rt][r] + bov) * vvv;
        }
#pragma unroll
        for (int rt = 0; rt < 4; ++rt)
#pragma unroll
            for (int r = 0; r < 8; ++r) {
                float v = sp[rt][r];                       // reduce 16 lanes (columns)
                v += __shfl_xor(v, 1, 32);
                v += __shfl_xor(v, 2, 32);
                v += __shfl_xor(v, 4, 32);
                v += __shfl_xor(v, 8, 32);
                if (l15 == 0) atomicAdd(&stile[rt * 16 + hsel + r], v);  // ds_add_f32
            }
        __syncthreads();
        if (tid < TT) scores_all[t0 + tid] = stile[tid];
        __syncthreads();
    }

    // ---- softmax over T=256 ----
    float s = scores_all[tid];
    float m = s;
    m = fmaxf(m, __shfl_xor(m, 16, 32));
    m = fmaxf(m, __shfl_xor(m, 8, 32));
    m = fmaxf(m, __shfl_xor(m, 4, 32));
    m = fmaxf(m, __shfl_xor(m, 2, 32));
    m = fmaxf(m, __shfl_xor(m, 1, 32));
    if (lane == 0) red[wave] = m;
    __syncthreads();
    if (tid == 0) {
        float mm = red[0];
#pragma unroll
        for (int w = 1; w < 8; ++w) mm = fmaxf(mm, red[w]);
        gscal[0] = mm;
    }
    __syncthreads();
    float e = __expf(s - gscal[0]);
    float l = e;
    l += __shfl_xor(l, 16, 32);
    l += __shfl_xor(l, 8, 32);
    l += __shfl_xor(l, 4, 32);
    l += __shfl_xor(l, 2, 32);
    l += __shfl_xor(l, 1, 32);
    if (lane == 0) red[wave] = l;
    __syncthreads();
    if (tid == 0) {
        float ss = red[0];
#pragma unroll
        for (int w = 1; w < 8; ++w) ss += red[w];
        gscal[1] = ss;
    }
    __syncthreads();
    float alpha = e / gscal[1];
    scores_all[tid] = alpha;
    out_alpha[b * TDIM + tid] = alpha;
    __syncthreads();

    // ---- s_t[b][n] = sum_t alpha[t] * h_history[b][t][n] (coalesced) ----
    float accv = 0.f;
    const float* hp = h_hist + (size_t)b * TDIM * HDIM + tid;
#pragma unroll 4
    for (int t = 0; t < TDIM; ++t) accv += scores_all[t] * hp[(size_t)t * HDIM];
    out_s[b * HDIM + tid] = accv;
}

// ---------------------------------------------------------------------------
extern "C" void kernel_launch(void* const* d_in, const int* in_sizes, int n_in,
                              void* d_out, int out_size, void* d_ws, size_t ws_size,
                              hipStream_t stream) {
    const float* h_tilde = (const float*)d_in[0];
    const float* c_t     = (const float*)d_in[1];
    const float* h_hist  = (const float*)d_in[2];
    const float* dT      = (const float*)d_in[3];
    const float* dL      = (const float*)d_in[4];
    const float* W_I     = (const float*)d_in[5];
    const float* W_Ih    = (const float*)d_in[6];
    const float* W_I1T   = (const float*)d_in[7];
    const float* W_I1L   = (const float*)d_in[8];
    const float* b_I     = (const float*)d_in[9];
    const float* W_oI    = (const float*)d_in[10];
    const float* b_oI    = (const float*)d_in[11];
    const float* v_t     = (const float*)d_in[12];

    char* ws = (char*)d_ws;
    unsigned short* wb1 = (unsigned short*)(ws);              // 128 KB
    unsigned short* wb2 = (unsigned short*)(ws + 131072);     // 128 KB
    float*          cp  = (float*)(ws + 262144);              // 512 KB

    float* out_s     = (float*)d_out;
    float* out_alpha = out_s + BDIM * HDIM;

    convert_weights<<<512, 256, 0, stream>>>(W_Ih, W_oI, wb1, wb2);
    current_proj_kernel<<<BDIM, HDIM, 0, stream>>>(h_tilde, c_t, W_I, b_I, cp);
    attn_fused<<<BDIM, 256, 0, stream>>>(h_hist, dT, dL, W_I1T, W_I1L, b_oI, v_t,
                                         wb1, wb2, cp, out_s, out_alpha);
}
